// ExpDModel_88338887344580
// MI455X (gfx1250) — matrix operand order repaired and checked by reading, not compiled, not run on hardware
//
#include <hip/hip_runtime.h>
#include <cstddef>

#define N_NODES 4096
#define N_EDGES 32768
#define DMODEL  768
#define HEADS   4
#define LAYERS  3
#define HD      (HEADS * DMODEL)
#define MAX_HOP 9
#define NEG_SLOPE 0.2f
#define LN_EPS  1e-5f

typedef __attribute__((ext_vector_type(16))) _Float16 v16h;
typedef __attribute__((ext_vector_type(8)))  _Float16 v8h;
typedef __attribute__((ext_vector_type(8)))  float    v8f;

#define HAS_TDM __has_builtin(__builtin_amdgcn_tensor_load_to_lds)

__device__ __forceinline__ float gelu_exact(float x) {
  return 0.5f * x * (1.0f + erff(x * 0.70710678118654752f));
}

// order-preserving float atomic max (init must be large-negative)
__device__ __forceinline__ void atomic_max_f32(float* addr, float val) {
  if (val >= 0.0f) atomicMax((int*)addr, __float_as_int(val));
  else             atomicMin((unsigned int*)addr, __float_as_uint(val));
}

// ---------------------------------------------------------------- fill
__global__ __launch_bounds__(256) void fill_kernel(float* __restrict__ p, float v, int n) {
  int i = blockIdx.x * 256 + threadIdx.x;
  if (i < n) p[i] = v;
}

// ---------------------------------------------------------------- structure injection
__global__ __launch_bounds__(256)
void inject_kernel(const float* __restrict__ x_text, const float* __restrict__ type_emb,
                   const float* __restrict__ hop_emb, const int* __restrict__ node_type,
                   const int* __restrict__ hop_id, float* __restrict__ x,
                   _Float16* __restrict__ x16) {
  int i = blockIdx.x * 256 + threadIdx.x;
  if (i >= N_NODES * DMODEL) return;
  int n = i / DMODEL;
  int d = i - n * DMODEL;
  int t  = node_type[n];
  int hp = hop_id[n];
  hp = hp < 0 ? 0 : (hp > MAX_HOP ? MAX_HOP : hp);
  float v = x_text[i] + type_emb[t * DMODEL + d] + hop_emb[hp * DMODEL + d];
  x[i] = v;
  x16[i] = (_Float16)v;
}

// ---------------------------------------------------------------- f32 -> f16 convert
__global__ __launch_bounds__(256)
void cvt_f16_kernel(const float* __restrict__ src, _Float16* __restrict__ dst, int n) {
  int i = blockIdx.x * 256 + threadIdx.x;
  if (i < n) dst[i] = (_Float16)src[i];
}

// ---------------------------------------------------------------- weight transpose + convert
// Wt[n][k] = (f16) W[k][n].  32x32 LDS tile keeps both sides coalesced.
__global__ __launch_bounds__(256)
void transpose_cvt_kernel(const float* __restrict__ W, _Float16* __restrict__ Wt,
                          int K, int Nt) {
  __shared__ float t[32][33];
  int bx = blockIdx.x * 32;              // along Nt
  int by = blockIdx.y * 32;              // along K
  int tx = threadIdx.x & 31;
  int ty = threadIdx.x >> 5;             // 32x8
  for (int j = ty; j < 32; j += 8)
    t[j][tx] = W[(size_t)(by + j) * Nt + bx + tx];
  __syncthreads();
  for (int j = ty; j < 32; j += 8)
    Wt[(size_t)(bx + j) * K + by + tx] = (_Float16)t[tx][j];
}

// ---------------------------------------------------------------- WMMA GEMM
// C[M,Nt] = A_f16[M,K] @ Bt_f16[Nt,K]^T (+ bias).  f16 multiply, f32 accumulate.
// 128x128x32 block tile, 8 waves (4 along M x 2 along N), wave tile 32x64.
// A/B tiles are identical 128x32 f16 2D tiles -> moved by the Tensor Data Mover
// into double-buffered LDS (pad 16B per 64B row => 80B LDS row stride).
#define BLK_M 128
#define BLK_N 128
#define BLK_K 32
#define APAD  8

#if HAS_TDM
typedef __attribute__((ext_vector_type(4))) unsigned tdm_v4u;
typedef __attribute__((ext_vector_type(8))) int      tdm_v8i;
typedef __attribute__((ext_vector_type(4))) int      tdm_v4i;

// 2D TDM tile load: 128 rows x 32 f16, LDS rows padded 64B->80B (D# per ISA 8.3/8.4).
// rem_d0/rem_d1 = remaining tensor extent from the tile origin (OOB rows read zero).
__device__ __forceinline__ void tdm_load_tile_f16(unsigned lds_addr, const _Float16* gaddr,
                                                  unsigned rem_d0, unsigned rem_d1,
                                                  unsigned stride_d0) {
  unsigned long long ga = (unsigned long long)gaddr;
  tdm_v4u g0;
  g0[0] = 1u;                                          // count=1, user descriptor
  g0[1] = lds_addr;                                    // lds_addr[31:0]
  g0[2] = (unsigned)(ga & 0xFFFFFFFFull);              // global_addr[31:0]
  g0[3] = (unsigned)((ga >> 32) & 0x1FFFFFFull) | 0x80000000u;  // addr[56:32] | type=2
  tdm_v8i g1;
  g1[0] = (int)((1u << 16) | (1u << 20) | (3u << 22) | (3u << 25));
  //        data_size=2B | pad_enable | pad_interval=16 DW | pad_amount=4 DW
  g1[1] = (int)((rem_d0 & 0xFFFFu) << 16);                               // tensor_dim0[15:0]
  g1[2] = (int)(((rem_d0 >> 16) & 0xFFFFu) | ((rem_d1 & 0xFFFFu) << 16));// dim0[31:16]|dim1[15:0]
  g1[3] = (int)(((rem_d1 >> 16) & 0xFFFFu) | ((unsigned)BLK_K << 16));   // dim1[31:16]|tile_dim0
  g1[4] = (int)BLK_M;                                                    // tile_dim1 (tile_dim2=0)
  g1[5] = (int)stride_d0;                                                // dim0_stride[31:0]
  g1[6] = 0;                                                             // stride[47:32], dim1_stride=0
  g1[7] = 0;
  tdm_v4i z4 = {0, 0, 0, 0};
#if __clang_major__ >= 23
  tdm_v8i z8 = {0, 0, 0, 0, 0, 0, 0, 0};
  __builtin_amdgcn_tensor_load_to_lds(g0, g1, z4, z4, z8, 0);
#else
  __builtin_amdgcn_tensor_load_to_lds(g0, g1, z4, z4, 0);
#endif
}
#else
// fallback staging: plain b128 copies of both f16 tiles
__device__ __forceinline__ void stage_pair_f16(const _Float16* __restrict__ Ablk,
                                               const _Float16* __restrict__ Bblk,
                                               int rowsA, int K, int k0,
                                               _Float16 (*as)[BLK_K + APAD],
                                               _Float16 (*bs)[BLK_K + APAD], int tid) {
  int r  = tid >> 2;
  int c8 = (tid & 3) * 8;
  for (int p = 0; p < 2; ++p) {
    int rr = r + p * 64;
    v8h va, vb;
    if (rr < rowsA) {
      va = *(const v8h*)(Ablk + (size_t)rr * K + k0 + c8);
    } else {
      for (int i = 0; i < 8; ++i) va[i] = (_Float16)0.0f;
    }
    vb = *(const v8h*)(Bblk + (size_t)rr * K + k0 + c8);
    *(v8h*)&as[rr][c8] = va;
    *(v8h*)&bs[rr][c8] = vb;
  }
}
#endif

__global__ __launch_bounds__(256)
void gemm_f16_wmma(const _Float16* __restrict__ A, const _Float16* __restrict__ Bt,
                   const float* __restrict__ bias, float* __restrict__ C,
                   int M, int K, int Nt) {
  __shared__ __align__(16) _Float16 As[2][BLK_M][BLK_K + APAD];
  __shared__ __align__(16) _Float16 Bs[2][BLK_N][BLK_K + APAD];

  const int tid     = threadIdx.x;
  const int lane    = tid & 31;
  const int wave    = tid >> 5;     // 0..7
  const int wm      = wave & 3;     // 4 waves along M
  const int wn      = wave >> 2;    // 2 waves along N
  const int halfsel = lane >> 4;    // K-split per ISA layout
  const int r16     = lane & 15;
  const int koffA   = halfsel * 8;  // A frag: lanes 0-15 K {0..7,16..23}; 16-31 {8..15,24..31}
  const int koffB   = halfsel * 8;  // B frag: lanes 0-15 K {0..7,16..23}; 16-31 {8..15,24..31}

  const int blockM = blockIdx.x * BLK_M;
  const int blockN = blockIdx.y * BLK_N;
  const _Float16* Ablk = A  + (size_t)blockM * K;
  const _Float16* Bblk = Bt + (size_t)blockN * K;

  v8f acc[2][4];
  for (int mi = 0; mi < 2; ++mi)
    for (int ni = 0; ni < 4; ++ni)
      for (int j = 0; j < 8; ++j) acc[mi][ni][j] = 0.0f;

  const int KT = K / BLK_K;

  // ---- prologue: buffer 0
#if HAS_TDM
  if (wave == 0) {
    tdm_load_tile_f16((unsigned)(unsigned long long)&As[0][0][0], Ablk,
                      (unsigned)K, (unsigned)(M - blockM), (unsigned)K);
    tdm_load_tile_f16((unsigned)(unsigned long long)&Bs[0][0][0], Bblk,
                      (unsigned)K, (unsigned)(Nt - blockN), (unsigned)K);
  }
#else
  stage_pair_f16(Ablk, Bblk, M - blockM, K, 0, As[0], Bs[0], tid);
#endif

  for (int kt = 0; kt < KT; ++kt) {
    const int cur = kt & 1;
#if HAS_TDM
    __builtin_amdgcn_s_wait_tensorcnt(0);   // no-op for waves with TENSORcnt==0
#endif
    __syncthreads();

    if (kt + 1 < KT) {
      const int k1 = (kt + 1) * BLK_K;
#if HAS_TDM
      if (wave == 0) {
        tdm_load_tile_f16((unsigned)(unsigned long long)&As[cur ^ 1][0][0], Ablk + k1,
                          (unsigned)(K - k1), (unsigned)(M - blockM), (unsigned)K);
        tdm_load_tile_f16((unsigned)(unsigned long long)&Bs[cur ^ 1][0][0], Bblk + k1,
                          (unsigned)(K - k1), (unsigned)(Nt - blockN), (unsigned)K);
      }
#else
      stage_pair_f16(Ablk, Bblk, M - blockM, K, k1, As[cur ^ 1], Bs[cur ^ 1], tid);
#endif
    }

    // ---- fragments (ISA 7.12.2 layouts), two b128 DS reads each
    v16h a[2], b[4];
    for (int mi = 0; mi < 2; ++mi) {
      int row = wm * 32 + mi * 16 + r16;
      v8h lo = *(const v8h*)&As[cur][row][koffA];
      v8h hi = *(const v8h*)&As[cur][row][16 + koffA];
      for (int i = 0; i < 8; ++i) { a[mi][i] = lo[i]; a[mi][i + 8] = hi[i]; }
    }
    for (int ni = 0; ni < 4; ++ni) {
      int col = wn * 64 + ni * 16 + r16;
      v8h lo = *(const v8h*)&Bs[cur][col][koffB];
      v8h hi = *(const v8h*)&Bs[cur][col][16 + koffB];
      for (int i = 0; i < 8; ++i) { b[ni][i] = lo[i]; b[ni][i + 8] = hi[i]; }
    }

    for (int mi = 0; mi < 2; ++mi)
      for (int ni = 0; ni < 4; ++ni)
        acc[mi][ni] = __builtin_amdgcn_wmma_f32_16x16x32_f16(
            false, a[mi], false, b[ni], (short)0, acc[mi][ni], false, false);

    __syncthreads();
  }

  // ---- store (C layout: VGPR j -> row halfsel*8+j, col = lane&15)
  for (int mi = 0; mi < 2; ++mi) {
    for (int ni = 0; ni < 4; ++ni) {
      int col = blockN + wn * 64 + ni * 16 + r16;
      float bv = bias ? bias[col] : 0.0f;
      for (int j = 0; j < 8; ++j) {
        int row = blockM + wm * 32 + mi * 16 + halfsel * 8 + j;
        if (row < M) C[(size_t)row * Nt + col] = acc[mi][ni][j] + bv;
      }
    }
  }
}

// ---------------------------------------------------------------- edge logits + segment max
__global__ __launch_bounds__(256)
void edge_logit_kernel(const float* __restrict__ xl, const float* __restrict__ xr,
                       const float* __restrict__ ee_rel, const float* __restrict__ att,
                       const int* __restrict__ src, const int* __restrict__ dst,
                       const int* __restrict__ rel, float* __restrict__ logit,
                       float* __restrict__ segmax) {
  int w    = blockIdx.x * 8 + (threadIdx.x >> 5);
  int lane = threadIdx.x & 31;
  int e = w >> 2;       // / HEADS
  int h = w & 3;        // % HEADS
  if (e >= N_EDGES) return;
  int s = src[e], d = dst[e], r = rel[e];
  const float* pl = xl     + (size_t)s * HD + h * DMODEL;
  const float* pr = xr     + (size_t)d * HD + h * DMODEL;
  const float* pe = ee_rel + (size_t)r * HD + h * DMODEL;
  const float* pa = att    + h * DMODEL;
  float acc = 0.0f;
  for (int i = lane; i < DMODEL; i += 32) {
    float v = pl[i] + pr[i] + pe[i];
    v = v > 0.0f ? v : NEG_SLOPE * v;
    acc += v * pa[i];
  }
  for (int m = 16; m; m >>= 1) acc += __shfl_xor(acc, m, 32);
  if (lane == 0) {
    logit[e * HEADS + h] = acc;
    atomic_max_f32(&segmax[d * HEADS + h], acc);
  }
}

// ---------------------------------------------------------------- exp + segment sum
__global__ __launch_bounds__(256)
void edge_exp_kernel(float* __restrict__ logit, const float* __restrict__ segmax,
                     const int* __restrict__ dst, float* __restrict__ den) {
  int i = blockIdx.x * 256 + threadIdx.x;
  if (i >= N_EDGES * HEADS) return;
  int e = i >> 2;
  int h = i & 3;
  int d = dst[e];
  float ex = expf(logit[i] - segmax[d * HEADS + h]);
  logit[i] = ex;
  atomicAdd(&den[d * HEADS + h], ex);
}

// ---------------------------------------------------------------- weighted scatter-add
__global__ __launch_bounds__(256)
void edge_scatter_kernel(const float* __restrict__ xl, const float* __restrict__ ex,
                         const float* __restrict__ den, const int* __restrict__ src,
                         const int* __restrict__ dst, float* __restrict__ agg) {
  int e = blockIdx.x;
  __shared__ float w[HEADS];
  int s = src[e], d = dst[e];
  if (threadIdx.x < HEADS)
    w[threadIdx.x] = ex[e * HEADS + threadIdx.x] / den[d * HEADS + threadIdx.x];
  __syncthreads();
  const float* px = xl + (size_t)s * HD;
  float* pa = agg + (size_t)d * HD;
  for (int i = threadIdx.x; i < HD; i += 256)
    atomicAdd(&pa[i], px[i] * w[i / DMODEL]);
}

// ---------------------------------------------------------------- head mean + GELU + residual
__global__ __launch_bounds__(256)
void node_update_kernel(const float* __restrict__ agg, const float* __restrict__ gnn_bias,
                        float* __restrict__ x, _Float16* __restrict__ x16) {
  int i = blockIdx.x * 256 + threadIdx.x;
  if (i >= N_NODES * DMODEL) return;
  int n = i / DMODEL;
  int d = i - n * DMODEL;
  const float* pa = agg + (size_t)n * HD + d;
  float s = 0.25f * (pa[0] + pa[DMODEL] + pa[2 * DMODEL] + pa[3 * DMODEL]);
  float out = s + gnn_bias[d];
  float xn = gelu_exact(out) + x[i];
  x[i] = xn;
  x16[i] = (_Float16)xn;
}

// ---------------------------------------------------------------- gated fusion (wave per node)
__global__ __launch_bounds__(256)
void gate_fuse_kernel(const float* __restrict__ x_text, const float* __restrict__ x,
                      const float* __restrict__ Wg, const float* __restrict__ bg,
                      float* __restrict__ xfused, _Float16* __restrict__ xf16) {
  int n    = blockIdx.x * 8 + (threadIdx.x >> 5);
  int lane = threadIdx.x & 31;
  if (n >= N_NODES) return;
  const float* pt = x_text + (size_t)n * DMODEL;
  const float* px = x      + (size_t)n * DMODEL;
  float acc = 0.0f;
  for (int i = lane; i < DMODEL; i += 32)
    acc += pt[i] * Wg[i] + px[i] * Wg[DMODEL + i];
  for (int m = 16; m; m >>= 1) acc += __shfl_xor(acc, m, 32);
  float g = 1.0f / (1.0f + expf(-(acc + bg[0])));
  for (int i = lane; i < DMODEL; i += 32) {
    float f = g * px[i] + (1.0f - g) * pt[i];
    xfused[(size_t)n * DMODEL + i] = f;
    xf16[(size_t)n * DMODEL + i]   = (_Float16)f;
  }
}

// ---------------------------------------------------------------- LayerNorm + GELU (block per row)
__global__ __launch_bounds__(256)
void ln_gelu_kernel(const float* __restrict__ h, const float* __restrict__ g,
                    const float* __restrict__ b, float* __restrict__ out) {
  int n = blockIdx.x;
  const float* p = h + (size_t)n * DMODEL;
  float s = 0.0f, ss = 0.0f;
  for (int i = threadIdx.x; i < DMODEL; i += 256) {
    float v = p[i];
    s += v;
    ss += v * v;
  }
  for (int m = 16; m; m >>= 1) { s += __shfl_xor(s, m, 32); ss += __shfl_xor(ss, m, 32); }
  __shared__ float rs[8], rss[8];
  int wv = threadIdx.x >> 5, lane = threadIdx.x & 31;
  if (lane == 0) { rs[wv] = s; rss[wv] = ss; }
  __syncthreads();
  if (threadIdx.x == 0) {
    float ts = 0.0f, tss = 0.0f;
    for (int i = 0; i < 8; ++i) { ts += rs[i]; tss += rss[i]; }
    rs[0] = ts; rss[0] = tss;
  }
  __syncthreads();
  float mean = rs[0] * (1.0f / DMODEL);
  float var  = rss[0] * (1.0f / DMODEL) - mean * mean;
  float inv  = rsqrtf(var + LN_EPS);
  for (int i = threadIdx.x; i < DMODEL; i += 256) {
    float v = (p[i] - mean) * inv * g[i] + b[i];
    out[(size_t)n * DMODEL + i] = gelu_exact(v);
  }
}

// ================================================================ host
extern "C" void kernel_launch(void* const* d_in, const int* in_sizes, int n_in,
                              void* d_out, int out_size, void* d_ws, size_t ws_size,
                              hipStream_t stream) {
  (void)in_sizes; (void)n_in; (void)out_size; (void)ws_size;
  const float* x_text   = (const float*)d_in[0];
  const float* type_emb = (const float*)d_in[1];
  const float* hop_emb  = (const float*)d_in[2];
  const float* rel_emb  = (const float*)d_in[3];
  const float* Wl       = (const float*)d_in[4];
  const float* bl       = (const float*)d_in[5];
  const float* Wr       = (const float*)d_in[6];
  const float* br       = (const float*)d_in[7];
  const float* We       = (const float*)d_in[8];
  const float* att      = (const float*)d_in[9];
  const float* gnn_bias = (const float*)d_in[10];
  const float* Wg       = (const float*)d_in[11];
  const float* bg       = (const float*)d_in[12];
  const float* Wp       = (const float*)d_in[13];
  const float* bp       = (const float*)d_in[14];
  const float* ln_g     = (const float*)d_in[15];
  const float* ln_b     = (const float*)d_in[16];
  const int* node_type  = (const int*)d_in[17];
  const int* hop_id     = (const int*)d_in[18];
  const int* edge_index = (const int*)d_in[19];
  const int* edge_attr  = (const int*)d_in[20];
  const int* esrc = edge_index;
  const int* edst = edge_index + N_EDGES;

  char* w = (char*)d_ws;
  size_t off = 0;
  auto take = [&](size_t bytes) -> void* {
    void* p = w + off;
    off = (off + bytes + 255) & ~(size_t)255;
    return p;
  };
  const size_t ND  = (size_t)N_NODES * DMODEL;
  const size_t WSZ = (size_t)DMODEL * HD;         // one weight matrix, elements
  float*    x      = (float*)take(ND * 4);
  _Float16* x16    = (_Float16*)take(ND * 2);     // f16 GEMM A input (reused for fused)
  float*    xl     = (float*)take((size_t)N_NODES * HD * 4);
  float*    xr     = (float*)take((size_t)N_NODES * HD * 4);
  float*    agg    = (float*)take((size_t)N_NODES * HD * 4);
  float*    ee_rel = (float*)take((size_t)64 * HD * 4);
  _Float16* rel16  = (_Float16*)take((size_t)64 * DMODEL * 2);
  float*    logit  = (float*)take((size_t)N_EDGES * HEADS * 4);
  float*    segmax = (float*)take((size_t)N_NODES * HEADS * 4);
  float*    den    = (float*)take((size_t)N_NODES * HEADS * 4);
  float*    xfused = (float*)take(ND * 4);
  float*    hproj  = (float*)take(ND * 4);
  _Float16* wtL    = (_Float16*)take(LAYERS * WSZ * 2);  // transposed f16 weights [Nt][K]
  _Float16* wtR    = (_Float16*)take(LAYERS * WSZ * 2);
  _Float16* wtE    = (_Float16*)take(LAYERS * WSZ * 2);
  _Float16* wtP    = (_Float16*)take((size_t)DMODEL * DMODEL * 2);

  const int nd_blocks = (int)((ND + 255) / 256);

  // ---- weight prep: transpose + f32->f16 (amortized; removes in-loop cvt + halves L2 traffic)
  dim3 gT(HD / 32, DMODEL / 32);
  for (int l = 0; l < LAYERS; ++l) {
    transpose_cvt_kernel<<<gT, 256, 0, stream>>>(Wl + (size_t)l * WSZ, wtL + (size_t)l * WSZ,
                                                 DMODEL, HD);
    transpose_cvt_kernel<<<gT, 256, 0, stream>>>(Wr + (size_t)l * WSZ, wtR + (size_t)l * WSZ,
                                                 DMODEL, HD);
    transpose_cvt_kernel<<<gT, 256, 0, stream>>>(We + (size_t)l * WSZ, wtE + (size_t)l * WSZ,
                                                 DMODEL, HD);
  }
  dim3 gTP(DMODEL / 32, DMODEL / 32);
  transpose_cvt_kernel<<<gTP, 256, 0, stream>>>(Wp, wtP, DMODEL, DMODEL);

  inject_kernel<<<nd_blocks, 256, 0, stream>>>(x_text, type_emb, hop_emb,
                                               node_type, hop_id, x, x16);
  cvt_f16_kernel<<<(64 * DMODEL + 255) / 256, 256, 0, stream>>>(rel_emb, rel16, 64 * DMODEL);

  for (int l = 0; l < LAYERS; ++l) {
    const _Float16* wtL_l = wtL + (size_t)l * WSZ;
    const _Float16* wtR_l = wtR + (size_t)l * WSZ;
    const _Float16* wtE_l = wtE + (size_t)l * WSZ;
    const float* bl_l = bl + (size_t)l * HD;
    const float* br_l = br + (size_t)l * HD;
    const float* at_l = att + (size_t)l * HEADS * DMODEL;
    const float* gb_l = gnn_bias + (size_t)l * DMODEL;

    dim3 gN(N_NODES / BLK_M, HD / BLK_N);   // 32 x 24
    gemm_f16_wmma<<<gN, 256, 0, stream>>>(x16, wtL_l, bl_l, xl, N_NODES, DMODEL, HD);
    gemm_f16_wmma<<<gN, 256, 0, stream>>>(x16, wtR_l, br_l, xr, N_NODES, DMODEL, HD);
    dim3 gE(1, HD / BLK_N);                 // 64-row A tile; TDM OOB rows read zero
    gemm_f16_wmma<<<gE, 256, 0, stream>>>(rel16, wtE_l, nullptr, ee_rel, 64, DMODEL, HD);

    int nh = N_NODES * HEADS;
    fill_kernel<<<(nh + 255) / 256, 256, 0, stream>>>(segmax, -3.0e38f, nh);
    fill_kernel<<<(nh + 255) / 256, 256, 0, stream>>>(den, 0.0f, nh);
    int nagg = N_NODES * HD;
    fill_kernel<<<(nagg + 255) / 256, 256, 0, stream>>>(agg, 0.0f, nagg);

    edge_logit_kernel<<<(N_EDGES * HEADS) / 8, 256, 0, stream>>>(
        xl, xr, ee_rel, at_l, esrc, edst, edge_attr, logit, segmax);
    edge_exp_kernel<<<(N_EDGES * HEADS) / 256, 256, 0, stream>>>(logit, segmax, edst, den);
    edge_scatter_kernel<<<N_EDGES, 256, 0, stream>>>(xl, logit, den, esrc, edst, agg);
    node_update_kernel<<<nd_blocks, 256, 0, stream>>>(agg, gb_l, x, x16);
  }

  gate_fuse_kernel<<<N_NODES / 8, 256, 0, stream>>>(x_text, x, Wg, bg, xfused, x16);
  dim3 gP(N_NODES / BLK_M, DMODEL / BLK_N); // 32 x 6
  gemm_f16_wmma<<<gP, 256, 0, stream>>>(x16, wtP, bp, hproj, N_NODES, DMODEL, DMODEL);
  ln_gelu_kernel<<<N_NODES, 256, 0, stream>>>(hproj, ln_g, ln_b, (float*)d_out);
}
